// RelayTokenGeometricConsistencyReranker_69793218560187
// MI455X (gfx1250) — matrix operand order repaired and checked by reading, not compiled, not run on hardware
//
#include <hip/hip_runtime.h>
#include <math.h>

// ---------------- problem constants ----------------
#define BB   64
#define N0c  768
#define N1c  384
#define Cc   256
#define K0c  512
#define K1c  256
#define Dc   768
#define PITERS 20

typedef __attribute__((ext_vector_type(16))) _Float16 v16h;
typedef __attribute__((ext_vector_type(8)))  float    v8f;
typedef __attribute__((ext_vector_type(8)))  _Float16 h8;
typedef __attribute__((ext_vector_type(4)))  float    f4;

// ============================================================
// 1) per-batch top-K selection (bitonic sort of (attn,idx)) +
//    gather/transform of centers
// ============================================================
__global__ __launch_bounds__(256)
void topk_kernel(const float* __restrict__ attn, const float* __restrict__ cen,
                 const int* __restrict__ num, const float* __restrict__ ss,
                 int* __restrict__ idxk, float* __restrict__ cenk,
                 int N, int K, int P) {
  __shared__ float sv[1024];
  __shared__ int   si[1024];
  int b = blockIdx.x;
  int nv = num[b];
  for (int e = threadIdx.x; e < P; e += 256) {
    float a = -1.0f;
    if (e < N) a = (e < nv) ? attn[(size_t)b * N + e] : 0.0f;
    sv[e] = a; si[e] = e;
  }
  __syncthreads();
  for (int k = 2; k <= P; k <<= 1) {
    for (int j = k >> 1; j > 0; j >>= 1) {
      for (int i = threadIdx.x; i < P; i += 256) {
        int ixj = i ^ j;
        if (ixj > i) {
          float va = sv[i], vb = sv[ixj];
          int ia = si[i], ib = si[ixj];
          bool aG = (va > vb) || (va == vb && ia < ib); // "a ranks higher"
          bool up = ((i & k) == 0);                     // up region = descending
          if (up ? !aG : aG) {
            sv[i] = vb; sv[ixj] = va;
            si[i] = ib; si[ixj] = ia;
          }
        }
      }
      __syncthreads();
    }
  }
  // write top-K indices + transformed centers
  float sx = ss[b * 4 + 0], sy = ss[b * 4 + 1], sz = ss[b * 4 + 2], sc = ss[b * 4 + 3];
  for (int e = threadIdx.x; e < K; e += 256) {
    int src = si[e];
    idxk[(size_t)b * K + e] = src;
    float cx = cen[((size_t)b * N + src) * 3 + 0];
    float cy = cen[((size_t)b * N + src) * 3 + 1];
    float cz = cen[((size_t)b * N + src) * 3 + 2];
    if (src < nv) { cx = cx * sc + sx; cy = cy * sc + sy; cz = cz * sc + sz; }
    cenk[((size_t)b * K + e) * 3 + 0] = cx;
    cenk[((size_t)b * K + e) * 3 + 1] = cy;
    cenk[((size_t)b * K + e) * 3 + 2] = cz;
  }
}

// ============================================================
// 2) WMMA feature GEMM: F[b,k,:] = rt[b, idxk[b,k], :] @ W_in + b_in
//    (B*K) x 256 @ 256 x 256, f16 in, fp32 acc, f16 out
// ============================================================
__global__ __launch_bounds__(512)
void feat_gemm_kernel(const float* __restrict__ rt, const int* __restrict__ idxk,
                      const float* __restrict__ W, const float* __restrict__ bias,
                      _Float16* __restrict__ F, int N, int K, int kshift) {
  __shared__ _Float16 As[64][40];
  __shared__ _Float16 Bs[32][72];
  int tid = threadIdx.x;
  int warp = tid >> 5, lane = tid & 31;
  int wi = warp >> 2, wj = warp & 3;
  int row0 = blockIdx.x * 64;
  int col0 = blockIdx.y * 64;
  int tm0 = wi * 16, tn0 = wj * 16;
  int m = lane & 15, h = lane >> 4;
  v8f acc = {};
  for (int kc = 0; kc < 256; kc += 32) {
    for (int e = tid; e < 64 * 32; e += 512) {
      int r = e >> 5, c = e & 31;
      int g = row0 + r;
      int b  = g >> kshift;
      int kk = g & (K - 1);
      int src = idxk[(size_t)b * K + kk];
      As[r][c] = (_Float16)rt[((size_t)b * N + src) * 256 + kc + c];
    }
    for (int e = tid; e < 32 * 64; e += 512) {
      int r = e >> 6, c = e & 63;
      Bs[r][c] = (_Float16)W[(size_t)(kc + r) * 256 + col0 + c];
    }
    __syncthreads();
    v16h af, bf;
#pragma unroll
    for (int e = 0; e < 16; ++e) {
      int kA = 16 * h + e;                              // 16-bit A 16x32 layout (contiguous k)
      af[e] = As[tm0 + m][kA];
      int kBv = 16 * h + e;                             // 16-bit B 32x16 layout
      bf[e] = Bs[kBv][tn0 + m];
    }
    acc = __builtin_amdgcn_wmma_f32_16x16x32_f16(false, af, false, bf,
                                                 (short)0, acc, false, false);
    __syncthreads();
  }
#pragma unroll
  for (int r = 0; r < 8; ++r) {
    int gm = row0 + tm0 + r + 8 * h;
    int gn = col0 + tn0 + m;
    float val = acc[r] + bias[gn];
    F[(size_t)gm * 256 + gn] = (_Float16)val;
  }
}

// ============================================================
// 3) squared row norms of F
// ============================================================
__global__ __launch_bounds__(256)
void rownorm_kernel(const _Float16* __restrict__ F, float* __restrict__ fnorm) {
  int warp = threadIdx.x >> 5, lane = threadIdx.x & 31;
  int row = blockIdx.x * 8 + warp;
  const h8* r = (const h8*)(F + (size_t)row * 256);
  h8 x = r[lane];
  float s = 0.0f;
#pragma unroll
  for (int t = 0; t < 8; ++t) { float xf = (float)x[t]; s += xf * xf; }
#pragma unroll
  for (int off = 16; off >= 1; off >>= 1) s += __shfl_xor(s, off);
  if (lane == 0) fnorm[row] = s;
}

// ============================================================
// 4) WMMA cross GEMM with fused nearest-neighbor argmin.
//    Fn tile staged via CDNA5 async global->LDS (ASYNCcnt).
//    d2[i][j] = |Fa_i|^2 + |Fn_j|^2 - 2 Fa_i.Fn_j ; match[i]=argmin_j
// ============================================================
__global__ __launch_bounds__(256)
void cross_argmin_kernel(const _Float16* __restrict__ F, const float* __restrict__ fnorm,
                         const int* __restrict__ anc, const int* __restrict__ pos,
                         const int* __restrict__ neg, int* __restrict__ match, int K) {
  __shared__ __align__(16) _Float16 Ns[16][264]; // 16 x 256 (+8 pad); row pitch 528B = 33*16
  int b = blockIdx.z, nsel = blockIdx.y;
  int ba = anc[b];
  int bn = (nsel == 0) ? pos[b] : neg[b];
  int warp = threadIdx.x >> 5, lane = threadIdx.x & 31;
  int i0 = blockIdx.x * 128 + warp * 16;
  int m = lane & 15, h = lane >> 4;

  v16h afr[8];
  const _Float16* Fa = F + ((size_t)ba * K + i0) * 256;
#pragma unroll
  for (int kc = 0; kc < 8; ++kc)
#pragma unroll
    for (int e = 0; e < 16; ++e) {
      int kk = 16 * h + e;
      afr[kc][e] = Fa[(size_t)m * 256 + kc * 32 + kk];
    }
  float fa2v[8];
#pragma unroll
  for (int r = 0; r < 8; ++r) fa2v[r] = fnorm[(size_t)ba * K + i0 + r + 8 * h];

  float bestV[8]; int bestJ[8];
#pragma unroll
  for (int r = 0; r < 8; ++r) { bestV[r] = 3.4e38f; bestJ[r] = 0; }

  const _Float16* Fn = F + (size_t)bn * K * 256;
  for (int j0 = 0; j0 < K; j0 += 16) {
    // async-stage the 16x256 f16 Fn tile: 512 x 16B chunks, 2 per thread
#pragma unroll
    for (int q = 0; q < 2; ++q) {
      int e = threadIdx.x + q * 256;       // chunk id 0..511
      int r = e >> 5;                      // 32 chunks per row
      int c0 = (e & 31) * 8;               // 8 halves = 16B per chunk
      unsigned ldsaddr = (unsigned)(uintptr_t)&Ns[r][c0];
      const _Float16* gp = Fn + (size_t)(j0 + r) * 256 + c0;
      asm volatile("global_load_async_to_lds_b128 %0, %1, off"
                   :: "v"(ldsaddr), "v"(gp) : "memory");
    }
    asm volatile("s_wait_asynccnt 0" ::: "memory");
    __syncthreads();
    v8f acc = {};
#pragma unroll
    for (int kc = 0; kc < 8; ++kc) {
      v16h bf;
#pragma unroll
      for (int e = 0; e < 16; ++e)
        bf[e] = Ns[m][kc * 32 + 16 * h + e]; // B[k][n] = Fn[j0+n][c]
      acc = __builtin_amdgcn_wmma_f32_16x16x32_f16(false, afr[kc], false, bf,
                                                   (short)0, acc, false, false);
    }
    float fn2 = fnorm[(size_t)bn * K + j0 + m];
    int j = j0 + m;
#pragma unroll
    for (int r = 0; r < 8; ++r) {
      float d2 = fa2v[r] + fn2 - 2.0f * acc[r];
      if (d2 < bestV[r] || (d2 == bestV[r] && j < bestJ[r])) { bestV[r] = d2; bestJ[r] = j; }
    }
    __syncthreads();
  }
#pragma unroll
  for (int r = 0; r < 8; ++r) {
    float v = bestV[r]; int jj = bestJ[r];
#pragma unroll
    for (int off = 8; off >= 1; off >>= 1) {
      float ov = __shfl_xor(v, off);
      int   oj = __shfl_xor(jj, off);
      if (ov < v || (ov == v && oj < jj)) { v = ov; jj = oj; }
    }
    if (m == 0)
      match[((size_t)b * 2 + nsel) * K + i0 + r + 8 * h] = jj;
  }
}

// ============================================================
// 5) build affinity matrix M (f16, fits the 192MB L2)
// ============================================================
__global__ __launch_bounds__(256)
void build_m_kernel(const float* __restrict__ cenk, const int* __restrict__ match,
                    const int* __restrict__ anc, const int* __restrict__ pos,
                    const int* __restrict__ neg, const int* __restrict__ num,
                    _Float16* __restrict__ Mm, int K) {
  __shared__ float ai[32][3], mi[32][3], aj[32][3], mj[32][3];
  int p = blockIdx.z; int b = p >> 1, nsel = p & 1;
  int ba = anc[b];
  int bn = (nsel == 0) ? pos[b] : neg[b];
  int i0 = blockIdx.x * 32, j0 = blockIdx.y * 32;
  const int* mt = match + (size_t)p * K;
  if (threadIdx.x < 32) {
    int t = threadIdx.x;
    int mii = mt[i0 + t], mjj = mt[j0 + t];
#pragma unroll
    for (int c = 0; c < 3; ++c) {
      ai[t][c] = cenk[((size_t)ba * K + i0 + t) * 3 + c];
      aj[t][c] = cenk[((size_t)ba * K + j0 + t) * 3 + c];
      mi[t][c] = cenk[((size_t)bn * K + mii) * 3 + c];
      mj[t][c] = cenk[((size_t)bn * K + mjj) * 3 + c];
    }
  }
  __syncthreads();
  int numA = num[ba], numN = num[bn];
#pragma unroll
  for (int q = 0; q < 4; ++q) {
    int e = threadIdx.x + q * 256;
    int ii = e >> 5, jj = e & 31;
    float dx = ai[ii][0] - aj[jj][0], dy = ai[ii][1] - aj[jj][1], dz = ai[ii][2] - aj[jj][2];
    float da = sqrtf(fmaxf(dx * dx + dy * dy + dz * dz, 1e-9f));
    float ex = mi[ii][0] - mj[jj][0], ey = mi[ii][1] - mj[jj][1], ez = mi[ii][2] - mj[jj][2];
    float dn = sqrtf(fmaxf(ex * ex + ey * ey + ez * ez, 1e-9f));
    float diff = da - dn;
    float mv = fmaxf(1.0f - diff * diff * (1.0f / 25.0f), 0.0f);
    int gi = i0 + ii, gj = j0 + jj;
    bool valid = (gi < numA) && (gj < numN) && (gi != gj);
    Mm[((size_t)p * K + gi) * K + gj] = valid ? (_Float16)mv : (_Float16)0.0f;
  }
}

// ============================================================
// 6) power iteration: y = M v ; v = y / (||y|| + 1e-8)
// ============================================================
__global__ __launch_bounds__(256)
void initv_kernel(float* __restrict__ v, int K) {
  int p = blockIdx.x;
  float val = rsqrtf((float)K);
  for (int e = threadIdx.x; e < K; e += 256) v[(size_t)p * K + e] = val;
}

__global__ __launch_bounds__(256)
void matvec_kernel(const _Float16* __restrict__ Mm, const float* __restrict__ v,
                   float* __restrict__ y, int K) {
  __shared__ __align__(16) float vs[512];
  int p = blockIdx.y;
  for (int e = threadIdx.x; e < K; e += 256) vs[e] = v[(size_t)p * K + e];
  __syncthreads();
  int warp = threadIdx.x >> 5, lane = threadIdx.x & 31;
  int row = blockIdx.x * 8 + warp;
  const _Float16* Mr = Mm + ((size_t)p * K + row) * K;
  float s = 0.0f;
  for (int c0 = 0; c0 < K; c0 += 256) {
    int c = c0 + lane * 8;
    h8 mv = *(const h8*)(Mr + c);        // global_load_b128
    f4 va0 = *(const f4*)&vs[c];         // ds_load_b128
    f4 va1 = *(const f4*)&vs[c + 4];
#pragma unroll
    for (int t = 0; t < 4; ++t) s += (float)mv[t] * va0[t];
#pragma unroll
    for (int t = 0; t < 4; ++t) s += (float)mv[t + 4] * va1[t];
  }
#pragma unroll
  for (int off = 16; off >= 1; off >>= 1) s += __shfl_xor(s, off);
  if (lane == 0) y[(size_t)p * K + row] = s;
}

__global__ __launch_bounds__(256)
void normalize_kernel(const float* __restrict__ y, float* __restrict__ v, int K) {
  __shared__ float part[8];
  int p = blockIdx.x;
  float s = 0.0f;
  for (int e = threadIdx.x; e < K; e += 256) { float x = y[(size_t)p * K + e]; s += x * x; }
#pragma unroll
  for (int off = 16; off >= 1; off >>= 1) s += __shfl_xor(s, off);
  int warp = threadIdx.x >> 5, lane = threadIdx.x & 31;
  if (lane == 0) part[warp] = s;
  __syncthreads();
  float tot = 0.0f;
#pragma unroll
  for (int w = 0; w < 8; ++w) tot += part[w];
  float inv = 1.0f / (sqrtf(tot) + 1e-8f);
  for (int e = threadIdx.x; e < K; e += 256) v[(size_t)p * K + e] = y[(size_t)p * K + e] * inv;
}

// ============================================================
// 7) sort v descending (bitonic) into v_all columns [colOff, colOff+K)
// ============================================================
__global__ __launch_bounds__(256)
void sortv_kernel(const float* __restrict__ v, float* __restrict__ vall, int K, int colOff) {
  __shared__ float sv[512];
  int p = blockIdx.x;
  for (int e = threadIdx.x; e < K; e += 256) sv[e] = v[(size_t)p * K + e];
  __syncthreads();
  for (int k = 2; k <= K; k <<= 1) {
    for (int j = k >> 1; j > 0; j >>= 1) {
      for (int i = threadIdx.x; i < K; i += 256) {
        int ixj = i ^ j;
        if (ixj > i) {
          float a = sv[i], c = sv[ixj];
          bool up = ((i & k) == 0);       // descending overall
          bool sw = up ? (a < c) : (a > c);
          if (sw) { sv[i] = c; sv[ixj] = a; }
        }
      }
      __syncthreads();
    }
  }
  for (int e = threadIdx.x; e < K; e += 256)
    vall[(size_t)p * Dc + colOff + e] = sv[e];
}

// ============================================================
// 8) final MLP: h = relu(v_all @ W1 + b1); scores = sigmoid(h @ W2 + b2)
// ============================================================
__global__ __launch_bounds__(256)
void mlp1_kernel(const float* __restrict__ va, const float* __restrict__ W1,
                 const float* __restrict__ b1, float* __restrict__ hbuf) {
  int r = blockIdx.x;
  int c = blockIdx.y * 256 + threadIdx.x;
  const float* vr = va + (size_t)r * Dc;
  float s = b1[c];
  for (int k = 0; k < Dc; ++k) s += vr[k] * W1[(size_t)k * Dc + c];
  hbuf[(size_t)r * Dc + c] = fmaxf(s, 0.0f);
}

__global__ __launch_bounds__(256)
void score_kernel(const float* __restrict__ hbuf, const float* __restrict__ W2,
                  const float* __restrict__ b2, float* __restrict__ out) {
  int warp = threadIdx.x >> 5, lane = threadIdx.x & 31;
  int row = blockIdx.x * 8 + warp;   // 0..127
  float s = 0.0f;
  for (int c = lane; c < Dc; c += 32) s += hbuf[(size_t)row * Dc + c] * W2[c];
#pragma unroll
  for (int off = 16; off >= 1; off >>= 1) s += __shfl_xor(s, off);
  if (lane == 0) {
    float sc = 1.0f / (1.0f + expf(-(s + b2[0])));
    out[row] = sc;
    out[128 + row] = ((row & 1) == 0) ? 1.0f : 0.0f; // targets[:,0]=1
  }
}

// ============================================================
// host launcher
// ============================================================
extern "C" void kernel_launch(void* const* d_in, const int* in_sizes, int n_in,
                              void* d_out, int out_size, void* d_ws, size_t ws_size,
                              hipStream_t stream) {
  const float* rt0  = (const float*)d_in[0];
  const float* rt1  = (const float*)d_in[1];
  const float* cen0 = (const float*)d_in[2];
  const float* cen1 = (const float*)d_in[3];
  const float* at0  = (const float*)d_in[4];
  const float* at1  = (const float*)d_in[5];
  const float* ss   = (const float*)d_in[6];
  const float* Win  = (const float*)d_in[7];
  const float* bin  = (const float*)d_in[8];
  const float* W1   = (const float*)d_in[9];
  const float* b1   = (const float*)d_in[10];
  const float* W2   = (const float*)d_in[11];
  const float* b2p  = (const float*)d_in[12];
  const int* anc  = (const int*)d_in[13];
  const int* pos  = (const int*)d_in[14];
  const int* neg  = (const int*)d_in[15];
  const int* num0 = (const int*)d_in[16];
  const int* num1 = (const int*)d_in[17];
  float* out = (float*)d_out;

  // workspace bump allocator (~86 MB total)
  size_t off = 0;
  auto alloc = [&](size_t bytes) -> void* {
    off = (off + 255) & ~(size_t)255;
    void* p = (char*)d_ws + off;
    off += bytes;
    return p;
  };
  const int Kmax = K0c;
  int*      idxk  = (int*)     alloc((size_t)BB * Kmax * 4);
  float*    cenk  = (float*)   alloc((size_t)BB * Kmax * 3 * 4);
  _Float16* F     = (_Float16*)alloc((size_t)BB * Kmax * Cc * 2);
  float*    fnorm = (float*)   alloc((size_t)BB * Kmax * 4);
  int*      match = (int*)     alloc((size_t)BB * 2 * Kmax * 4);
  _Float16* Mm    = (_Float16*)alloc((size_t)BB * 2 * Kmax * Kmax * 2);
  float*    v     = (float*)   alloc((size_t)BB * 2 * Kmax * 4);
  float*    y     = (float*)   alloc((size_t)BB * 2 * Kmax * 4);
  float*    vall  = (float*)   alloc((size_t)BB * 2 * Dc * 4);
  float*    hbuf  = (float*)   alloc((size_t)BB * 2 * Dc * 4);

  for (int d = 0; d < 2; ++d) {
    const float* rt   = d == 0 ? rt0  : rt1;
    const float* cen  = d == 0 ? cen0 : cen1;
    const float* attn = d == 0 ? at0  : at1;
    const int*   num  = d == 0 ? num0 : num1;
    int N = d == 0 ? N0c : N1c;
    int K = d == 0 ? K0c : K1c;
    int P = d == 0 ? 1024 : 512;
    int kshift = d == 0 ? 9 : 8;
    int colOff = d == 0 ? 0 : K0c;

    topk_kernel<<<BB, 256, 0, stream>>>(attn, cen, num, ss, idxk, cenk, N, K, P);
    feat_gemm_kernel<<<dim3(BB * K / 64, Cc / 64), 512, 0, stream>>>(
        rt, idxk, Win, bin, F, N, K, kshift);
    rownorm_kernel<<<BB * K / 8, 256, 0, stream>>>(F, fnorm);
    cross_argmin_kernel<<<dim3(K / 128, 2, BB), 256, 0, stream>>>(
        F, fnorm, anc, pos, neg, match, K);
    build_m_kernel<<<dim3(K / 32, K / 32, 2 * BB), 256, 0, stream>>>(
        cenk, match, anc, pos, neg, num, Mm, K);
    initv_kernel<<<2 * BB, 256, 0, stream>>>(v, K);
    for (int it = 0; it < PITERS; ++it) {
      matvec_kernel<<<dim3(K / 8, 2 * BB), 256, 0, stream>>>(Mm, v, y, K);
      normalize_kernel<<<2 * BB, 256, 0, stream>>>(y, v, K);
    }
    sortv_kernel<<<2 * BB, 256, 0, stream>>>(v, vall, K, colOff);
  }

  mlp1_kernel<<<dim3(2 * BB, Dc / 256), 256, 0, stream>>>(vall, W1, b1, hbuf);
  score_kernel<<<16, 256, 0, stream>>>(hbuf, W2, b2p, out);
}
